// MPNNModel_76974403879029
// MI455X (gfx1250) — matrix operand order repaired and checked by reading, not compiled, not run on hardware
//
#include <hip/hip_runtime.h>
#include <cstdint>
#include <cstddef>

#define Nn 32768
#define Ee 65536
#define Gg 1024
#define Dd 64
#define EHh 128
#define NIi 15
#define EIi 5
#define OUTo 12
#define MP_STEPS 6
#define S2S_STEPS 6
#define KP_PAD 16   // LDS row padding (bf16 elems) = 32B: keeps 32B alignment, spreads banks

typedef __bf16 bf16_t;
typedef bf16_t v16bf __attribute__((ext_vector_type(16)));
typedef bf16_t v8bf  __attribute__((ext_vector_type(8)));
typedef float  v8f   __attribute__((ext_vector_type(8)));

// ---------------- device helpers ----------------
__device__ __forceinline__ unsigned short f2bf(float f) {
  unsigned u = __float_as_uint(f);
  u += 0x7FFFu + ((u >> 16) & 1u);            // round-to-nearest-even
  return (unsigned short)(u >> 16);
}
__device__ __forceinline__ float bf2f(unsigned short b) {
  return __uint_as_float(((unsigned)b) << 16);
}
__device__ __forceinline__ float sigmoidf_(float x) { return 1.0f / (1.0f + __expf(-x)); }
// monotonic float->uint encoding for atomicMax-based segment max
__device__ __forceinline__ unsigned fenc(float f) {
  unsigned u = __float_as_uint(f);
  return (u & 0x80000000u) ? ~u : (u | 0x80000000u);
}
__device__ __forceinline__ float fdec(unsigned k) {
  unsigned u = (k & 0x80000000u) ? (k & 0x7FFFFFFFu) : ~k;
  return __uint_as_float(u);
}

// ---------------- generic fills / converts ----------------
__global__ void fill_f32_kernel(float* p, float v, int n) {
  int i = blockIdx.x * 256 + threadIdx.x;
  if (i < n) p[i] = v;
}
__global__ void fill_u32_kernel(unsigned* p, unsigned v, int n) {
  int i = blockIdx.x * 256 + threadIdx.x;
  if (i < n) p[i] = v;
}
__global__ void cvt_bf16_kernel(const float* __restrict__ s, unsigned short* __restrict__ d, int n) {
  int i = blockIdx.x * 256 + threadIdx.x;
  if (i < n) d[i] = f2bf(s[i]);
}
__global__ void copy_f32_kernel(const float* __restrict__ s, float* __restrict__ d, int n) {
  int i = blockIdx.x * 256 + threadIdx.x;
  if (i < n) d[i] = s[i];
}

// ---------------- WMMA GEMM: C[M,N] = act(A[M,K](bf16) @ W[N,K](bf16)^T + bias[N]) ----------------
// Block = 256 threads = 8 waves; block tile = 256 rows x 64 cols; wave tile = 32x64.
// B panel (64 cols x K, contiguous in W) is async-copied to LDS once per block via
// global_load_async_to_lds_b128 (ASYNCcnt), then read as fragments with ds_load.
// Requires M%256==0, N%64==0, K%32==0, K<=128.
__global__ __launch_bounds__(256) void wmma_gemm_bf16(
    const unsigned short* __restrict__ A,
    const unsigned short* __restrict__ Wt,
    const float* __restrict__ bias,
    float* __restrict__ Cf, unsigned short* __restrict__ Cb,
    int M, int N, int K, int relu)
{
  __shared__ unsigned short Bs[64 * (128 + KP_PAD)];
  const int tid  = threadIdx.x;
  const int lane = tid & 31;
  const int wave = tid >> 5;
  const int row0 = blockIdx.y * 256 + wave * 32;
  const int col0 = blockIdx.x * 64;
  const int KP   = K + KP_PAD;

  // ---- async-stage B panel: rows col0..col0+63 of W (contiguous 64*K bf16) -> LDS ----
  {
    const int ck = K >> 3;                       // 16B chunks per W row
    const int nchunk = 64 * ck;
    const unsigned long long sbase = (unsigned long long)(Wt + (size_t)col0 * K);
    for (int i = tid; i < nchunk; i += 256) {
      const int col = i / ck;
      const int c8  = (i - col * ck) << 3;       // element offset within row
      const unsigned voff = (unsigned)((col * K + c8) * 2);
      const unsigned ldsa = (unsigned)(size_t)&Bs[col * KP + c8];
      asm volatile("global_load_async_to_lds_b128 %0, %1, %2"
                   :: "v"(ldsa), "v"(voff), "s"(sbase) : "memory");
    }
    asm volatile("s_wait_asynccnt 0x0" ::: "memory");
  }
  __syncthreads();
  if (row0 >= M) return;

  const int hi = lane >> 4;     // K sub-block select per ISA A/B layouts
  const int lr = lane & 15;

  v8f zero = {0.f,0.f,0.f,0.f,0.f,0.f,0.f,0.f};
  v8f acc[2][4];
#pragma unroll
  for (int rt = 0; rt < 2; ++rt)
#pragma unroll
    for (int t = 0; t < 4; ++t) acc[rt][t] = zero;

  const unsigned short* Arow0 = A + (size_t)(row0 + lr) * K;
  const unsigned short* Arow1 = A + (size_t)(row0 + 16 + lr) * K;

  for (int k0 = 0; k0 < K; k0 += 32) {
    if (k0 + 32 < K) {
      __builtin_prefetch(Arow0 + k0 + 32, 0, 1);   // global_prefetch_b8
      __builtin_prefetch(Arow1 + k0 + 32, 0, 1);
    }
    // A fragments: elems 0..7 = K(k0+hi*8..+7), elems 8..15 = K(k0+hi*8+16..+23)
    v16bf a0, a1;
    {
      v8bf lo0 = *(const v8bf*)(Arow0 + k0 + hi * 8);
      v8bf hh0 = *(const v8bf*)(Arow0 + k0 + hi * 8 + 16);
      v8bf lo1 = *(const v8bf*)(Arow1 + k0 + hi * 8);
      v8bf hh1 = *(const v8bf*)(Arow1 + k0 + hi * 8 + 16);
#pragma unroll
      for (int i = 0; i < 8; ++i) {
        a0[i] = lo0[i]; a0[8 + i] = hh0[i];
        a1[i] = lo1[i]; a1[8 + i] = hh1[i];
      }
    }
#pragma unroll
    for (int t = 0; t < 4; ++t) {
      // B fragment from LDS: column (t*16+lr), elems 0..7 = K(k0+hi*8..+7), elems 8..15 = K(k0+hi*8+16..+23)
      const unsigned short* bp = &Bs[(t * 16 + lr) * KP + k0 + hi * 8];
      v8bf blo = *(const v8bf*)bp;
      v8bf bhi = *(const v8bf*)(bp + 16);
      v16bf bfrag;
#pragma unroll
      for (int i = 0; i < 8; ++i) { bfrag[i] = blo[i]; bfrag[8 + i] = bhi[i]; }
      acc[0][t] = __builtin_amdgcn_wmma_f32_16x16x32_bf16(
          false, a0, false, bfrag, (short)0, acc[0][t], false, false);
      acc[1][t] = __builtin_amdgcn_wmma_f32_16x16x32_bf16(
          false, a1, false, bfrag, (short)0, acc[1][t], false, false);
    }
  }

  // Epilogue: VGPR r / lane l -> (M = row0 + rt*16 + r + 8*hi, N = col0 + t*16 + lr)
#pragma unroll
  for (int rt = 0; rt < 2; ++rt) {
#pragma unroll
    for (int t = 0; t < 4; ++t) {
      const int colb = col0 + t * 16 + lr;
      const float bv = bias ? bias[colb] : 0.0f;
#pragma unroll
      for (int r = 0; r < 8; ++r) {
        const int row = row0 + rt * 16 + r + hi * 8;
        float v = acc[rt][t][r] + bv;
        if (relu) v = v > 0.0f ? v : 0.0f;
        const size_t idx = (size_t)row * N + colb;
        if (Cf) Cf[idx] = v;
        if (Cb) Cb[idx] = f2bf(v);
      }
    }
  }
}

// ---------------- small VALU kernels ----------------
// he = relu(e_feat[E,5] @ e1_w[128,5]^T + e1_b) -> bf16
__global__ void edge_mlp1_kernel(const float* __restrict__ ef, const float* __restrict__ w,
                                 const float* __restrict__ b, unsigned short* __restrict__ he) {
  const size_t idx = (size_t)blockIdx.x * 256 + threadIdx.x;   // E*128
  const int h = (int)(idx & 127);
  const size_t e = idx >> 7;
  if (e >= Ee) return;
  float s = b[h];
#pragma unroll
  for (int i = 0; i < EIi; ++i) s += ef[e * EIi + i] * w[h * EIi + i];
  he[idx] = f2bf(s > 0.f ? s : 0.f);
}

// out = relu(n_feat[N,15] @ lin0_w[64,15]^T + b) -> f32 + bf16
__global__ void lin0_kernel(const float* __restrict__ nf, const float* __restrict__ w,
                            const float* __restrict__ b, float* __restrict__ out,
                            unsigned short* __restrict__ hbf) {
  const size_t idx = (size_t)blockIdx.x * 256 + threadIdx.x;   // N*64
  const int j = (int)(idx & 63);
  const size_t n = idx >> 6;
  if (n >= Nn) return;
  float s = b[j];
#pragma unroll
  for (int i = 0; i < NIi; ++i) s += nf[n * NIi + i] * w[j * NIi + i];
  s = s > 0.f ? s : 0.f;
  out[idx] = s;
  hbf[idx] = f2bf(s);
}

// per-edge matvec + scatter: one wave per edge, lane owns cols (lane, lane+32)
__global__ __launch_bounds__(256) void msg_scatter_kernel(
    const float* __restrict__ out, const unsigned short* __restrict__ We,
    const int* __restrict__ src, const int* __restrict__ dst, float* __restrict__ agg) {
  const int lane = threadIdx.x & 31;
  const int wave = threadIdx.x >> 5;
  const size_t e = (size_t)blockIdx.x * 8 + wave;
  if (e >= Ee) return;
  const int s = src[e], d = dst[e];
  const unsigned short* W = We + e * (size_t)(Dd * Dd);
  const float* x = out + (size_t)s * Dd;
  float a0 = 0.f, a1 = 0.f;
#pragma unroll 4
  for (int k = 0; k < Dd; ++k) {
    const float xv = x[k];                       // uniform across wave -> broadcast
    a0 += xv * bf2f(W[k * Dd + lane]);
    a1 += xv * bf2f(W[k * Dd + 32 + lane]);
  }
  atomicAdd(&agg[(size_t)d * Dd + lane], a0);
  atomicAdd(&agg[(size_t)d * Dd + 32 + lane], a1);
}

__global__ void relu_bias_kernel(const float* __restrict__ agg, const float* __restrict__ cb,
                                 unsigned short* __restrict__ mbf) {
  const size_t idx = (size_t)blockIdx.x * 256 + threadIdx.x;   // N*64
  if (idx >= (size_t)Nn * Dd) return;
  const int j = (int)(idx & 63);
  float v = agg[idx] + cb[j];
  mbf[idx] = f2bf(v > 0.f ? v : 0.f);
}

// GRU cell (gate order r,z,n)
__global__ void gru_gate_kernel(const float* __restrict__ gi, const float* __restrict__ gh,
                                float* __restrict__ h, unsigned short* __restrict__ hbf) {
  const size_t idx = (size_t)blockIdx.x * 256 + threadIdx.x;   // N*64
  if (idx >= (size_t)Nn * Dd) return;
  const int j = (int)(idx & 63);
  const size_t n = idx >> 6;
  const size_t b = n * (size_t)(3 * Dd);
  const float r = sigmoidf_(gi[b + j] + gh[b + j]);
  const float z = sigmoidf_(gi[b + 64 + j] + gh[b + 64 + j]);
  const float ng = tanhf(gi[b + 128 + j] + r * gh[b + 128 + j]);
  const float hv = (1.f - z) * ng + z * h[idx];
  h[idx] = hv;
  hbf[idx] = f2bf(hv);
}

// LSTM cell (gate order i,f,g,o), gates = gA + gB
__global__ void lstm_cell_kernel(const float* __restrict__ gA, const float* __restrict__ gB,
                                 float* __restrict__ c, float* __restrict__ h,
                                 unsigned short* __restrict__ hbf) {
  const size_t idx = (size_t)blockIdx.x * 256 + threadIdx.x;   // G*64
  if (idx >= (size_t)Gg * Dd) return;
  const int j = (int)(idx & 63);
  const size_t g = idx >> 6;
  const size_t b = g * (size_t)(4 * Dd);
  const float ig = sigmoidf_(gA[b + j] + gB[b + j]);
  const float fg = sigmoidf_(gA[b + 64 + j] + gB[b + 64 + j]);
  const float gg = tanhf(gA[b + 128 + j] + gB[b + 128 + j]);
  const float og = sigmoidf_(gA[b + 192 + j] + gB[b + 192 + j]);
  const float cv = fg * c[idx] + ig * gg;
  c[idx] = cv;
  const float hv = og * tanhf(cv);
  h[idx] = hv;
  hbf[idx] = f2bf(hv);
}

__global__ void att_dot_kernel(const float* __restrict__ out, const float* __restrict__ q,
                               const int* __restrict__ gid, float* __restrict__ e) {
  const int n = blockIdx.x * 256 + threadIdx.x;
  if (n >= Nn) return;
  const int g = gid[n];
  float s = 0.f;
#pragma unroll 4
  for (int j = 0; j < Dd; ++j) s += out[(size_t)n * Dd + j] * q[(size_t)g * Dd + j];
  e[n] = s;
}

__global__ void seg_max_kernel(const float* __restrict__ e, const int* __restrict__ gid,
                               unsigned* __restrict__ menc) {
  const int n = blockIdx.x * 256 + threadIdx.x;
  if (n >= Nn) return;
  atomicMax(&menc[gid[n]], fenc(e[n]));
}

__global__ void seg_expsum_kernel(const float* __restrict__ e, const int* __restrict__ gid,
                                  const unsigned* __restrict__ menc, float* __restrict__ ex,
                                  float* __restrict__ s) {
  const int n = blockIdx.x * 256 + threadIdx.x;
  if (n >= Nn) return;
  const int g = gid[n];
  const float v = __expf(e[n] - fdec(menc[g]));
  ex[n] = v;
  atomicAdd(&s[g], v);
}

__global__ void att_scatter_kernel(const float* __restrict__ ex, const float* __restrict__ s,
                                   const int* __restrict__ gid, const float* __restrict__ out,
                                   float* __restrict__ read) {
  const size_t idx = (size_t)blockIdx.x * 256 + threadIdx.x;   // N*64
  if (idx >= (size_t)Nn * Dd) return;
  const int j = (int)(idx & 63);
  const size_t n = idx >> 6;
  const int g = gid[n];
  const float alpha = ex[n] / s[g];
  atomicAdd(&read[(size_t)g * Dd + j], alpha * out[idx]);
}

__global__ void qstar_kernel(const float* __restrict__ q, const float* __restrict__ read,
                             float* __restrict__ qstar, unsigned short* __restrict__ qbf) {
  const size_t idx = (size_t)blockIdx.x * 256 + threadIdx.x;   // G*128
  if (idx >= (size_t)Gg * 2 * Dd) return;
  const int j = (int)(idx & 127);
  const size_t g = idx >> 7;
  const float v = (j < Dd) ? q[g * Dd + j] : read[g * Dd + (j - Dd)];
  qstar[idx] = v;
  qbf[idx] = f2bf(v);
}

__global__ void lin2_kernel(const float* __restrict__ y, const float* __restrict__ w,
                            const float* __restrict__ b, float* __restrict__ pred) {
  const int idx = blockIdx.x * 256 + threadIdx.x;              // G*12
  if (idx >= Gg * OUTo) return;
  const int o = idx % OUTo;
  const int g = idx / OUTo;
  float s = b[o];
#pragma unroll 4
  for (int j = 0; j < Dd; ++j) s += y[(size_t)g * Dd + j] * w[o * Dd + j];
  pred[idx] = s;
}

// ---------------- host launcher ----------------
extern "C" void kernel_launch(void* const* d_in, const int* in_sizes, int n_in,
                              void* d_out, int out_size, void* d_ws, size_t ws_size,
                              hipStream_t stream) {
  (void)in_sizes; (void)n_in; (void)out_size;
  const float* n_feat = (const float*)d_in[0];
  const float* e_feat = (const float*)d_in[1];
  const int*   src    = (const int*)d_in[2];
  const int*   dst    = (const int*)d_in[3];
  const int*   gid    = (const int*)d_in[4];
  const float* lin0_w = (const float*)d_in[5];
  const float* lin0_b = (const float*)d_in[6];
  const float* e1_w   = (const float*)d_in[7];
  const float* e1_b   = (const float*)d_in[8];
  const float* e2_w   = (const float*)d_in[9];
  const float* e2_b   = (const float*)d_in[10];
  const float* conv_b = (const float*)d_in[11];
  const float* gru_wih = (const float*)d_in[12];
  const float* gru_whh = (const float*)d_in[13];
  const float* gru_bih = (const float*)d_in[14];
  const float* gru_bhh = (const float*)d_in[15];
  const float* lin1_w = (const float*)d_in[16];
  const float* lin1_b = (const float*)d_in[17];
  const float* lin2_w = (const float*)d_in[18];
  const float* lin2_b = (const float*)d_in[19];
  const float* lwih[3]; const float* lwhh[3]; const float* lbih[3]; const float* lbhh[3];
  for (int l = 0; l < 3; ++l) {
    lwih[l] = (const float*)d_in[20 + 4 * l];
    lwhh[l] = (const float*)d_in[21 + 4 * l];
    lbih[l] = (const float*)d_in[22 + 4 * l];
    lbhh[l] = (const float*)d_in[23 + 4 * l];
  }

  // ---- workspace layout (256B aligned slabs) ----
  size_t off = 0;
  auto alloc = [&](size_t bytes) -> void* {
    void* p = (char*)d_ws + off;
    off += (bytes + 255) & ~(size_t)255;
    return p;
  };
  unsigned short* We_bf   = (unsigned short*)alloc((size_t)Ee * Dd * Dd * 2);   // 512 MB
  unsigned short* he_bf   = (unsigned short*)alloc((size_t)Ee * EHh * 2);
  unsigned short* e2w_bf  = (unsigned short*)alloc((size_t)Dd * Dd * EHh * 2);
  unsigned short* gwih_bf = (unsigned short*)alloc((size_t)3 * Dd * Dd * 2);
  unsigned short* gwhh_bf = (unsigned short*)alloc((size_t)3 * Dd * Dd * 2);
  unsigned short* lwih_bf[3]; unsigned short* lwhh_bf[3];
  for (int l = 0; l < 3; ++l) {
    lwih_bf[l] = (unsigned short*)alloc((size_t)4 * Dd * (l == 0 ? 2 * Dd : Dd) * 2);
    lwhh_bf[l] = (unsigned short*)alloc((size_t)4 * Dd * Dd * 2);
  }
  unsigned short* lin1w_bf = (unsigned short*)alloc((size_t)Dd * 2 * Dd * 2);
  float* outf  = (float*)alloc((size_t)Nn * Dd * 4);
  unsigned short* h_bf = (unsigned short*)alloc((size_t)Nn * Dd * 2);
  float* agg   = (float*)alloc((size_t)Nn * Dd * 4);
  unsigned short* m_bf = (unsigned short*)alloc((size_t)Nn * Dd * 2);
  float* gi    = (float*)alloc((size_t)Nn * 3 * Dd * 4);
  float* gh    = (float*)alloc((size_t)Nn * 3 * Dd * 4);
  float* qstar = (float*)alloc((size_t)Gg * 2 * Dd * 4);
  unsigned short* qstar_bf = (unsigned short*)alloc((size_t)Gg * 2 * Dd * 2);
  float* hs[3]; float* cs[3]; unsigned short* hbf_l[3];
  for (int l = 0; l < 3; ++l) {
    hs[l] = (float*)alloc((size_t)Gg * Dd * 4);
    cs[l] = (float*)alloc((size_t)Gg * Dd * 4);
    hbf_l[l] = (unsigned short*)alloc((size_t)Gg * Dd * 2);
  }
  float* gA    = (float*)alloc((size_t)Gg * 4 * Dd * 4);
  float* gB    = (float*)alloc((size_t)Gg * 4 * Dd * 4);
  float* e_att = (float*)alloc((size_t)Nn * 4);
  float* ex    = (float*)alloc((size_t)Nn * 4);
  unsigned* menc = (unsigned*)alloc((size_t)Gg * 4);
  float* ssum  = (float*)alloc((size_t)Gg * 4);
  float* readv = (float*)alloc((size_t)Gg * Dd * 4);
  float* y     = (float*)alloc((size_t)Gg * Dd * 4);
  if (off > ws_size || d_ws == nullptr) return;   // workspace too small: bail deterministically

  const dim3 B256(256);
  auto blocks = [](size_t n) { return dim3((unsigned)((n + 255) / 256)); };
  auto cvt = [&](const float* s, unsigned short* d, int n) {
    cvt_bf16_kernel<<<blocks((size_t)n), B256, 0, stream>>>(s, d, n);
  };
  auto gemm = [&](const unsigned short* A, const unsigned short* Wt, const float* bias,
                  float* Cf, unsigned short* Cb, int M, int N, int K, int relu) {
    dim3 g((unsigned)(N / 64), (unsigned)(M / 256));
    wmma_gemm_bf16<<<g, B256, 0, stream>>>(A, Wt, bias, Cf, Cb, M, N, K, relu);
  };

  // ---- weight conversions (f32 -> bf16) ----
  cvt(e2_w, e2w_bf, Dd * Dd * EHh);
  cvt(gru_wih, gwih_bf, 3 * Dd * Dd);
  cvt(gru_whh, gwhh_bf, 3 * Dd * Dd);
  for (int l = 0; l < 3; ++l) {
    cvt(lwih[l], lwih_bf[l], 4 * Dd * (l == 0 ? 2 * Dd : Dd));
    cvt(lwhh[l], lwhh_bf[l], 4 * Dd * Dd);
  }
  cvt(lin1_w, lin1w_bf, Dd * 2 * Dd);

  // ---- edge MLP: he = relu(e_feat@e1^T+b) ; W_e = he@e2^T + e2_b (WMMA, bf16 out) ----
  edge_mlp1_kernel<<<blocks((size_t)Ee * EHh), B256, 0, stream>>>(e_feat, e1_w, e1_b, he_bf);
  gemm(he_bf, e2w_bf, e2_b, nullptr, We_bf, Ee, Dd * Dd, EHh, 0);

  // ---- lin0 ----
  lin0_kernel<<<blocks((size_t)Nn * Dd), B256, 0, stream>>>(n_feat, lin0_w, lin0_b, outf, h_bf);

  // ---- message passing ----
  for (int step = 0; step < MP_STEPS; ++step) {
    fill_f32_kernel<<<blocks((size_t)Nn * Dd), B256, 0, stream>>>(agg, 0.f, Nn * Dd);
    msg_scatter_kernel<<<dim3(Ee / 8), B256, 0, stream>>>(outf, We_bf, src, dst, agg);
    relu_bias_kernel<<<blocks((size_t)Nn * Dd), B256, 0, stream>>>(agg, conv_b, m_bf);
    gemm(m_bf, gwih_bf, gru_bih, gi, nullptr, Nn, 3 * Dd, Dd, 0);
    gemm(h_bf, gwhh_bf, gru_bhh, gh, nullptr, Nn, 3 * Dd, Dd, 0);
    gru_gate_kernel<<<blocks((size_t)Nn * Dd), B256, 0, stream>>>(gi, gh, outf, h_bf);
  }

  // ---- Set2Set init ----
  fill_f32_kernel<<<blocks((size_t)Gg * 2 * Dd), B256, 0, stream>>>(qstar, 0.f, Gg * 2 * Dd);
  fill_u32_kernel<<<blocks((size_t)Gg * Dd), B256, 0, stream>>>((unsigned*)qstar_bf, 0u, Gg * Dd);
  for (int l = 0; l < 3; ++l) {
    fill_f32_kernel<<<blocks((size_t)Gg * Dd), B256, 0, stream>>>(hs[l], 0.f, Gg * Dd);
    fill_f32_kernel<<<blocks((size_t)Gg * Dd), B256, 0, stream>>>(cs[l], 0.f, Gg * Dd);
    fill_u32_kernel<<<blocks((size_t)Gg * Dd / 2), B256, 0, stream>>>((unsigned*)hbf_l[l], 0u, Gg * Dd / 2);
  }

  // ---- Set2Set steps ----
  for (int step = 0; step < S2S_STEPS; ++step) {
    const unsigned short* x = qstar_bf;
    int in_dim = 2 * Dd;
    for (int l = 0; l < 3; ++l) {
      gemm(x, lwih_bf[l], lbih[l], gA, nullptr, Gg, 4 * Dd, in_dim, 0);
      gemm(hbf_l[l], lwhh_bf[l], lbhh[l], gB, nullptr, Gg, 4 * Dd, Dd, 0);
      lstm_cell_kernel<<<blocks((size_t)Gg * Dd), B256, 0, stream>>>(gA, gB, cs[l], hs[l], hbf_l[l]);
      x = hbf_l[l];
      in_dim = Dd;
    }
    att_dot_kernel<<<blocks((size_t)Nn), B256, 0, stream>>>(outf, hs[2], gid, e_att);
    fill_u32_kernel<<<blocks((size_t)Gg), B256, 0, stream>>>(menc, 0x00800000u /* enc(-FLT_MAX) */, Gg);
    fill_f32_kernel<<<blocks((size_t)Gg), B256, 0, stream>>>(ssum, 0.f, Gg);
    fill_f32_kernel<<<blocks((size_t)Gg * Dd), B256, 0, stream>>>(readv, 0.f, Gg * Dd);
    seg_max_kernel<<<blocks((size_t)Nn), B256, 0, stream>>>(e_att, gid, menc);
    seg_expsum_kernel<<<blocks((size_t)Nn), B256, 0, stream>>>(e_att, gid, menc, ex, ssum);
    att_scatter_kernel<<<blocks((size_t)Nn * Dd), B256, 0, stream>>>(ex, ssum, gid, outf, readv);
    qstar_kernel<<<blocks((size_t)Gg * 2 * Dd), B256, 0, stream>>>(hs[2], readv, qstar, qstar_bf);
  }

  // ---- head: y = relu(q_star @ lin1^T + b) ; pred = y @ lin2^T + b ----
  gemm(qstar_bf, lin1w_bf, lin1_b, y, nullptr, Gg, Dd, 2 * Dd, 1);
  float* pred = (float*)d_out;
  lin2_kernel<<<blocks((size_t)Gg * OUTo), B256, 0, stream>>>(y, lin2_w, lin2_b, pred);
  copy_f32_kernel<<<blocks((size_t)Gg * 2 * Dd), B256, 0, stream>>>(qstar, pred + Gg * OUTo, Gg * 2 * Dd);
}
